// PrawnMKPModel_6287832121565
// MI455X (gfx1250) — compile-verified, matrix operand order checked by reading, not run on hardware
//
#include <hip/hip_runtime.h>
#include <math.h>

typedef __attribute__((ext_vector_type(16))) _Float16 v16h;
typedef __attribute__((ext_vector_type(8)))  float    v8f;

// ---------------------------------------------------------------------------
// wave32 reductions
// ---------------------------------------------------------------------------
__device__ __forceinline__ float wave_sum(float v) {
#pragma unroll
  for (int o = 16; o > 0; o >>= 1) v += __shfl_down(v, o, 32);
  return __shfl(v, 0, 32);
}
__device__ __forceinline__ float wave_max(float v) {
#pragma unroll
  for (int o = 16; o > 0; o >>= 1) v = fmaxf(v, __shfl_down(v, o, 32));
  return __shfl(v, 0, 32);
}

// ---------------------------------------------------------------------------
// CDNA5 16-bit fragment layout: lane&15 = matrix row (A) / col (B);
// lanes 0-15 hold K = {k0..k0+7, k0+16..k0+23}, lanes 16-31 hold
// K = {k0+8..k0+15, k0+24..k0+31}.  kbase = k0 + ((lane&16)?8:0).
// 4x b128 loads, branch-free, packed f32->f16 converts.
// ---------------------------------------------------------------------------
__device__ __forceinline__ v16h frag_vec(const float* __restrict__ p, int kbase) {
  const float4 a = *(const float4*)(p + kbase);
  const float4 b = *(const float4*)(p + kbase + 4);
  const float4 c = *(const float4*)(p + kbase + 16);
  const float4 d = *(const float4*)(p + kbase + 20);
  v16h f;
  f[0]  = (_Float16)a.x; f[1]  = (_Float16)a.y; f[2]  = (_Float16)a.z; f[3]  = (_Float16)a.w;
  f[4]  = (_Float16)b.x; f[5]  = (_Float16)b.y; f[6]  = (_Float16)b.z; f[7]  = (_Float16)b.w;
  f[8]  = (_Float16)c.x; f[9]  = (_Float16)c.y; f[10] = (_Float16)c.z; f[11] = (_Float16)c.w;
  f[12] = (_Float16)d.x; f[13] = (_Float16)d.y; f[14] = (_Float16)d.z; f[15] = (_Float16)d.w;
  return f;
}

#define WMMA_F16(a, b, c) \
  __builtin_amdgcn_wmma_f32_16x16x32_f16(false, (a), false, (b), (short)0, (c), false, false)

// Epilogue for one 16x16 accumulator tile at (rowBase, colBase).
template <bool RES, bool GELU>
__device__ __forceinline__ void store_tile(float* __restrict__ Cb, const v8f& acc,
                                           int rowBase, int colBase, int lane, int ldc,
                                           const float* __restrict__ bias,
                                           const float* __restrict__ ResB, int ldr,
                                           float alpha) {
  const int col = colBase + (lane & 15);
  const float bv = bias ? bias[col] : 0.0f;
  const int r0 = rowBase + ((lane & 16) >> 1);  // +8 for upper half-lanes
#pragma unroll
  for (int r = 0; r < 8; ++r) {
    const int row = r0 + r;
    float v = alpha * acc[r] + bv;
    if (RES) v += ResB[(long long)row * ldr + col];
    if (GELU) v = 0.5f * v * (1.0f + erff(v * 0.70710678118654752f));
    Cb[(long long)row * ldc + col] = v;
  }
}

// ---------------------------------------------------------------------------
// Aligned batched GEMM: C[bz] = act( alpha*A[bz] x B[bz]^T + bias (+Res) )
// A: [M x K] row-major, B: [N x K] row-major (i.e. x @ W.T).
// REQUIRES K % 32 == 0, M % 32 == 0, N % 32 == 0.
// One wave computes a 32x32 macro-tile: 2 A-frags x 2 B-frags -> 4 WMMAs
// per K-step (2x operand reuse, 4x amortized address math).
// ---------------------------------------------------------------------------
template <bool RES, bool GELU>
__global__ void __launch_bounds__(32)
gemm32_wmma_f16(const float* __restrict__ A, int lda, long long sA,
                const float* __restrict__ Bm, int ldb, long long sB,
                float* __restrict__ C, int ldc, long long sC,
                const float* __restrict__ bias,
                const float* __restrict__ Res, int ldr, long long sR,
                int M, int N, int K, float alpha)
{
  const int lane = threadIdx.x & 31;
  const int tN = blockIdx.x;
  const int tM = blockIdx.y;
  const int bz = blockIdx.z;

  const int r0 = tM * 32 + (lane & 15);
  const int c0 = tN * 32 + (lane & 15);
  const float* Ap0 = A  + (long long)bz * sA + (long long)r0 * lda;
  const float* Ap1 = Ap0 + (long long)16 * lda;
  const float* Bp0 = Bm + (long long)bz * sB + (long long)c0 * ldb;
  const float* Bp1 = Bp0 + (long long)16 * ldb;
  const int kh = (lane & 16) ? 8 : 0;

  v8f acc00 = {}, acc01 = {}, acc10 = {}, acc11 = {};
  for (int k0 = 0; k0 < K; k0 += 32) {
    __builtin_prefetch(Ap0 + k0 + 128, 0, 1);   // speculative, drop-on-fault
    __builtin_prefetch(Ap1 + k0 + 128, 0, 1);
    __builtin_prefetch(Bp0 + k0 + 128, 0, 1);
    __builtin_prefetch(Bp1 + k0 + 128, 0, 1);
    const int kb = k0 + kh;
    const v16h a0 = frag_vec(Ap0, kb);
    const v16h a1 = frag_vec(Ap1, kb);
    const v16h b0 = frag_vec(Bp0, kb);
    const v16h b1 = frag_vec(Bp1, kb);
    acc00 = WMMA_F16(a0, b0, acc00);
    acc01 = WMMA_F16(a0, b1, acc01);
    acc10 = WMMA_F16(a1, b0, acc10);
    acc11 = WMMA_F16(a1, b1, acc11);
  }

  float* Cb = C + (long long)bz * sC;
  const float* ResB = RES ? (Res + (long long)bz * sR) : nullptr;
  const int rb = tM * 32, cb = tN * 32;
  store_tile<RES, GELU>(Cb, acc00, rb,      cb,      lane, ldc, bias, ResB, ldr, alpha);
  store_tile<RES, GELU>(Cb, acc01, rb,      cb + 16, lane, ldc, bias, ResB, ldr, alpha);
  store_tile<RES, GELU>(Cb, acc10, rb + 16, cb,      lane, ldc, bias, ResB, ldr, alpha);
  store_tile<RES, GELU>(Cb, acc11, rb + 16, cb + 16, lane, ldc, bias, ResB, ldr, alpha);
}

// ---------------------------------------------------------------------------
// Edge GEMM (attention): one wave per 16x16 tile; K % 32 == 0 still required.
// OOB rows/cols are clamped on load (garbage feeds only unstored outputs);
// stores are guarded.
// ---------------------------------------------------------------------------
__global__ void __launch_bounds__(32)
gemm16e_wmma_f16(const float* __restrict__ A, int lda, long long sA,
                 const float* __restrict__ Bm, int ldb, long long sB,
                 float* __restrict__ C, int ldc, long long sC,
                 int M, int N, int K, float alpha)
{
  const int lane = threadIdx.x & 31;
  const int tN = blockIdx.x;
  const int tM = blockIdx.y;
  const int bz = blockIdx.z;

  const int arow = tM * 16 + (lane & 15);
  const int bcol = tN * 16 + (lane & 15);
  const int ar = arow < M ? arow : M - 1;
  const int bc = bcol < N ? bcol : N - 1;

  const float* Ap = A  + (long long)bz * sA + (long long)ar * lda;
  const float* Bp = Bm + (long long)bz * sB + (long long)bc * ldb;
  const int kh = (lane & 16) ? 8 : 0;

  v8f acc = {};
  for (int k0 = 0; k0 < K; k0 += 32) {
    const v16h af = frag_vec(Ap, k0 + kh);
    const v16h bf = frag_vec(Bp, k0 + kh);
    acc = WMMA_F16(af, bf, acc);
  }

  float* Cb = C + (long long)bz * sC;
  const int col = bcol;
  if (col < N) {
    const int rbase = tM * 16 + ((lane & 16) >> 1);
#pragma unroll
    for (int r = 0; r < 8; ++r) {
      const int row = rbase + r;
      if (row < M) Cb[(long long)row * ldc + col] = alpha * acc[r];
    }
  }
}

// ---------------------------------------------------------------------------
// LayerNorm: one wave per row
// ---------------------------------------------------------------------------
__global__ void __launch_bounds__(32)
layernorm_k(const float* __restrict__ x, float* __restrict__ y,
            const float* __restrict__ w, const float* __restrict__ b, int d)
{
  const long long row = blockIdx.x;
  const int lane = threadIdx.x;
  const float* xr = x + row * d;
  float* yr = y + row * d;
  float s = 0.f;
  for (int i = lane; i < d; i += 32) s += xr[i];
  const float mean = wave_sum(s) / (float)d;
  float v = 0.f;
  for (int i = lane; i < d; i += 32) { float t = xr[i] - mean; v += t * t; }
  const float rstd = rsqrtf(wave_sum(v) / (float)d + 1e-5f);
  for (int i = lane; i < d; i += 32)
    yr[i] = (xr[i] - mean) * rstd * w[i] + b[i];
}

// ---------------------------------------------------------------------------
// Row softmax (in place): one wave per (b,h,q) row; stride Sc, valid len S.
// Pad columns [S, Sc) are zeroed so P x V can run with K = Sc (mult of 32).
// ---------------------------------------------------------------------------
__global__ void __launch_bounds__(32)
softmax_rows_k(float* __restrict__ s, int S, int Sc)
{
  float* p = s + (long long)blockIdx.x * Sc;
  const int lane = threadIdx.x;
  float m = -3.0e38f;
  for (int i = lane; i < S; i += 32) m = fmaxf(m, p[i]);
  m = wave_max(m);
  float sum = 0.f;
  for (int i = lane; i < S; i += 32) { float e = __expf(p[i] - m); p[i] = e; sum += e; }
  sum = wave_sum(sum);
  const float inv = 1.0f / sum;
  for (int i = lane; i < Sc; i += 32) p[i] = (i < S) ? p[i] * inv : 0.0f;
}

// ---------------------------------------------------------------------------
// im2col for 16x16 stride-16 conv: out[(b*196+p)*768 + (c*256+iy*16+ix)]
// ---------------------------------------------------------------------------
__global__ void im2col_k(const float* __restrict__ x, float* __restrict__ out, int total)
{
  int idx = blockIdx.x * blockDim.x + threadIdx.x;
  if (idx >= total) return;
  const int k = idx % 768;
  const int m = idx / 768;
  const int b = m / 196;
  const int p = m % 196;
  const int py = p / 14, px = p % 14;
  const int c = k >> 8;
  const int rem = k & 255;
  const int iy = rem >> 4, ix = rem & 15;
  out[idx] = x[(((long long)b * 3 + c) * 224 + (py * 16 + iy)) * 224 + (px * 16 + ix)];
}

__global__ void add_pos_k(float* __restrict__ t, const float* __restrict__ pos,
                          int total, int Dm, int NPm)
{
  int idx = blockIdx.x * blockDim.x + threadIdx.x;
  if (idx >= total) return;
  const int d = idx % Dm;
  const int p = (idx / Dm) % NPm;
  t[idx] += pos[(p + 1) * Dm + d];
}

// xe[b,0,:] = cls + pos[0]; xe[b,1+i,:] = ptok[b, unmasked[b,i], :]
__global__ void build_enc_in_k(const float* __restrict__ tok, const int* __restrict__ un,
                               const float* __restrict__ cls, const float* __restrict__ pos,
                               float* __restrict__ xe, int total)
{
  int idx = blockIdx.x * blockDim.x + threadIdx.x;
  if (idx >= total) return;
  const int d = idx % 384;
  const int s = (idx / 384) % 50;
  const int b = idx / (384 * 50);
  float v;
  if (s == 0) v = cls[d] + pos[d];
  else {
    const int p = un[b * 49 + (s - 1)];
    v = tok[((long long)b * 196 + p) * 384 + d];
  }
  xe[idx] = v;
}

// qkv [B,S,3D] -> q/k [B,H,S,dh], vT [B,H,dh,Sc] (Sc = K-padded seq stride)
__global__ void split_qkv_k(const float* __restrict__ qkv,
                            float* __restrict__ q, float* __restrict__ k,
                            float* __restrict__ vt,
                            int S, int Dm, int Hh, int dh, int Sc, int total)
{
  int idx = blockIdx.x * blockDim.x + threadIdx.x;
  if (idx >= total) return;
  const int j = idx % (3 * Dm);
  const int s = (idx / (3 * Dm)) % S;
  const int b = idx / (3 * Dm * S);
  const int which = j / Dm;
  const int d = j % Dm;
  const int h = d / dh;
  const int e = d % dh;
  const float v = qkv[idx];
  const long long bh = (long long)b * Hh + h;
  if (which == 0)      q [(bh * S + s) * dh + e] = v;
  else if (which == 1) k [(bh * S + s) * dh + e] = v;
  else                 vt[(bh * dh + e) * Sc + s] = v;
}

// [B,H,S,dh] -> [B,S,D]
__global__ void merge_heads_k(const float* __restrict__ ah, float* __restrict__ out,
                              int S, int Dm, int Hh, int dh, int total)
{
  int idx = blockIdx.x * blockDim.x + threadIdx.x;
  if (idx >= total) return;
  const int d = idx % Dm;
  const int s = (idx / Dm) % S;
  const int b = idx / (Dm * S);
  const int h = d / dh, e = d % dh;
  out[idx] = ah[(((long long)b * Hh + h) * S + s) * dh + e];
}

// full[b,p,:] = mask_token + dec_pos[1+p]   (vis positions overwritten next)
__global__ void dec_init_k(float* __restrict__ full, const float* __restrict__ mtok,
                           const float* __restrict__ dpos, int total)
{
  int idx = blockIdx.x * blockDim.x + threadIdx.x;
  if (idx >= total) return;
  const int d = idx % 256;
  const int p = (idx / 256) % 196;
  full[idx] = mtok[d] + dpos[(p + 1) * 256 + d];
}

// full[b, un[b,i], :] = xd[b, 1+i, :] + dec_pos[1+un[b,i]]
__global__ void dec_scatter_k(const float* __restrict__ xd, const int* __restrict__ un,
                              const float* __restrict__ dpos, float* __restrict__ full,
                              int total)
{
  int idx = blockIdx.x * blockDim.x + threadIdx.x;
  if (idx >= total) return;
  const int d = idx % 256;
  const int i = (idx / 256) % 49;
  const int b = idx / (256 * 49);
  const int p = un[b * 49 + i];
  full[((long long)b * 196 + p) * 256 + d] =
      xd[((long long)b * 50 + 1 + i) * 256 + d] + dpos[(p + 1) * 256 + d];
}

// ---------------------------------------------------------------------------
// Host side
// ---------------------------------------------------------------------------
static inline int cdiv(int a, int b) { return (a + b - 1) / b; }

extern "C" void kernel_launch(void* const* d_in, const int* in_sizes, int n_in,
                              void* d_out, int out_size, void* d_ws, size_t ws_size,
                              hipStream_t stream)
{
  (void)in_sizes; (void)n_in; (void)out_size; (void)ws_size;
  const float* x         = (const float*)d_in[0];
  const int*   unmasked  = (const int*)  d_in[1];
  /* d_in[2] masked_idx unused: dec_init covers all positions, vis overwrites */
  const float* conv_w    = (const float*)d_in[3];
  const float* conv_b    = (const float*)d_in[4];
  const float* cls_token = (const float*)d_in[5];
  const float* pos_embed = (const float*)d_in[6];
  const float* e_ln1_w = (const float*)d_in[7],  *e_ln1_b = (const float*)d_in[8];
  const float* e_qkv_w = (const float*)d_in[9],  *e_qkv_b = (const float*)d_in[10];
  const float* e_out_w = (const float*)d_in[11], *e_out_b = (const float*)d_in[12];
  const float* e_ln2_w = (const float*)d_in[13], *e_ln2_b = (const float*)d_in[14];
  const float* e_fc1_w = (const float*)d_in[15], *e_fc1_b = (const float*)d_in[16];
  const float* e_fc2_w = (const float*)d_in[17], *e_fc2_b = (const float*)d_in[18];
  const float* e_nw    = (const float*)d_in[19], *e_nb    = (const float*)d_in[20];
  const float* demb_w  = (const float*)d_in[21], *demb_b  = (const float*)d_in[22];
  const float* mask_tok= (const float*)d_in[23];
  const float* dpos    = (const float*)d_in[24];
  const float* dc_ln1_w= (const float*)d_in[25], *dc_ln1_b= (const float*)d_in[26];
  const float* dc_qkv_w= (const float*)d_in[27], *dc_qkv_b= (const float*)d_in[28];
  const float* dc_out_w= (const float*)d_in[29], *dc_out_b= (const float*)d_in[30];
  const float* dc_ln2_w= (const float*)d_in[31], *dc_ln2_b= (const float*)d_in[32];
  const float* dc_fc1_w= (const float*)d_in[33], *dc_fc1_b= (const float*)d_in[34];
  const float* dc_fc2_w= (const float*)d_in[35], *dc_fc2_b= (const float*)d_in[36];
  const float* dc_nw   = (const float*)d_in[37], *dc_nb   = (const float*)d_in[38];
  const float* pred_w  = (const float*)d_in[39], *pred_b  = (const float*)d_in[40];
  float* outp = (float*)d_out;

  const int Bc = 64, NPc = 196, Dc = 384, Se = 50, He = 6,  dhe = 64, DEPTHe = 6;
  const int DDc = 256, Sd = 196, Hd = 8,  dhd = 32, DEPTHd = 4;
  const int Sce = (Se + 31) & ~31;   // 64:  K-padded seq for encoder attention
  const int Scd = (Sd + 31) & ~31;   // 224: K-padded seq for decoder attention

  // ---- workspace carve (256-aligned) ----
  char* base = (char*)d_ws;
  size_t off = 0;
  auto alloc = [&](size_t floats) -> float* {
    float* p = (float*)(base + off);
    off += ((floats * 4) + 255) & ~(size_t)255;
    return p;
  };
  float* im2col = alloc((size_t)Bc * NPc * 768);
  float* ptok   = alloc((size_t)Bc * NPc * Dc);
  float* xe     = alloc((size_t)Bc * Se * Dc);
  float* lnt    = alloc((size_t)Bc * Sd * DDc);          // >= enc usage
  float* qkvb   = alloc((size_t)Bc * Sd * 3 * DDc);      // >= enc usage
  float* qh     = alloc((size_t)Bc * Hd * Sd * dhd);     // >= enc 64*6*50*64
  float* kh     = alloc((size_t)Bc * Hd * Sd * dhd);
  float* vth    = alloc((size_t)Bc * Hd * dhd * Scd);    // >= enc 64*6*64*64
  float* scores = alloc((size_t)Bc * Hd * Sd * Scd);     // >= enc 64*6*50*64
  float* attnh  = alloc((size_t)Bc * Hd * Sd * dhd);
  float* merged = alloc((size_t)Bc * Sd * DDc);
  float* hidden = alloc((size_t)Bc * Sd * 4 * DDc);
  float* xd     = alloc((size_t)Bc * Se * DDc);
  float* xfull  = alloc((size_t)Bc * NPc * DDc);

  // Aligned path (M,N % 32 == 0): 32x32 macro-tile kernel.
  auto gemm = [&](const float* A, int lda, long long sA,
                  const float* Bm, int ldb, long long sB,
                  float* C, int ldc, long long sC,
                  const float* bias, const float* Res, int ldr, long long sR,
                  int M, int N, int K, int batch, float alpha, int gelu) {
    dim3 g(N / 32, M / 32, batch);
    if (Res)
      gemm32_wmma_f16<true, false><<<g, dim3(32), 0, stream>>>(
          A, lda, sA, Bm, ldb, sB, C, ldc, sC, bias, Res, ldr, sR, M, N, K, alpha);
    else if (gelu)
      gemm32_wmma_f16<false, true><<<g, dim3(32), 0, stream>>>(
          A, lda, sA, Bm, ldb, sB, C, ldc, sC, bias, nullptr, 0, 0, M, N, K, alpha);
    else
      gemm32_wmma_f16<false, false><<<g, dim3(32), 0, stream>>>(
          A, lda, sA, Bm, ldb, sB, C, ldc, sC, bias, nullptr, 0, 0, M, N, K, alpha);
  };
  // Attention path (ragged M/N, no bias/act): 16x16 edge kernel.
  auto gemm_e = [&](const float* A, int lda, long long sA,
                    const float* Bm, int ldb, long long sB,
                    float* C, int ldc, long long sC,
                    int M, int N, int K, int batch, float alpha) {
    dim3 g(cdiv(N, 16), cdiv(M, 16), batch);
    gemm16e_wmma_f16<<<g, dim3(32), 0, stream>>>(
        A, lda, sA, Bm, ldb, sB, C, ldc, sC, M, N, K, alpha);
  };

  auto run_block = [&](float* xbuf, int S, int Sc, int Dm, int Hh, int dh,
                       const float* ln1w, const float* ln1b,
                       const float* qkvw, const float* qkvbias,
                       const float* outw, const float* outb,
                       const float* ln2w, const float* ln2b,
                       const float* fc1w, const float* fc1b,
                       const float* fc2w, const float* fc2b) {
    const int rows = Bc * S;
    const int BH = Bc * Hh;
    layernorm_k<<<rows, 32, 0, stream>>>(xbuf, lnt, ln1w, ln1b, Dm);
    gemm(lnt, Dm, 0, qkvw, Dm, 0, qkvb, 3 * Dm, 0, qkvbias,
         nullptr, 0, 0, rows, 3 * Dm, Dm, 1, 1.0f, 0);
    const int tq = rows * 3 * Dm;
    split_qkv_k<<<cdiv(tq, 256), 256, 0, stream>>>(qkvb, qh, kh, vth, S, Dm, Hh, dh, Sc, tq);
    // scores = (1/sqrt(dh)) * Q x K^T   [batch B*H], row stride Sc
    gemm_e(qh, dh, (long long)S * dh, kh, dh, (long long)S * dh,
           scores, Sc, (long long)S * Sc, S, S, dh, BH, 1.0f / sqrtf((float)dh));
    softmax_rows_k<<<BH * S, 32, 0, stream>>>(scores, S, Sc);
    // O = P x V   (K = Sc, multiple of 32; pad probs are zero)
    gemm_e(scores, Sc, (long long)S * Sc, vth, Sc, (long long)dh * Sc,
           attnh, dh, (long long)S * dh, S, dh, Sc, BH, 1.0f);
    const int tm = rows * Dm;
    merge_heads_k<<<cdiv(tm, 256), 256, 0, stream>>>(attnh, merged, S, Dm, Hh, dh, tm);
    // x = x + O @ Wo^T + bo   (residual fused)
    gemm(merged, Dm, 0, outw, Dm, 0, xbuf, Dm, 0, outb,
         xbuf, Dm, 0, rows, Dm, Dm, 1, 1.0f, 0);
    layernorm_k<<<rows, 32, 0, stream>>>(xbuf, lnt, ln2w, ln2b, Dm);
    // h = gelu(ln2 @ W1^T + b1)
    gemm(lnt, Dm, 0, fc1w, Dm, 0, hidden, 4 * Dm, 0, fc1b,
         nullptr, 0, 0, rows, 4 * Dm, Dm, 1, 1.0f, 1);
    // x = x + h @ W2^T + b2
    gemm(hidden, 4 * Dm, 0, fc2w, 4 * Dm, 0, xbuf, Dm, 0, fc2b,
         xbuf, Dm, 0, rows, Dm, 4 * Dm, 1, 1.0f, 0);
  };

  // ---- patch embed (conv as GEMM) ----
  const int tim = Bc * NPc * 768;
  im2col_k<<<cdiv(tim, 256), 256, 0, stream>>>(x, im2col, tim);
  gemm(im2col, 768, 0, conv_w, 768, 0, ptok, Dc, 0, conv_b,
       nullptr, 0, 0, Bc * NPc, Dc, 768, 1, 1.0f, 0);
  const int tpt = Bc * NPc * Dc;
  add_pos_k<<<cdiv(tpt, 256), 256, 0, stream>>>(ptok, pos_embed, tpt, Dc, NPc);

  // ---- gather visible + cls ----
  const int txe = Bc * Se * Dc;
  build_enc_in_k<<<cdiv(txe, 256), 256, 0, stream>>>(ptok, unmasked, cls_token, pos_embed, xe, txe);

  // ---- encoder ----
  for (int i = 0; i < DEPTHe; ++i)
    run_block(xe, Se, Sce, Dc, He, dhe,
              e_ln1_w + i * Dc, e_ln1_b + i * Dc,
              e_qkv_w + (size_t)i * 3 * Dc * Dc, e_qkv_b + i * 3 * Dc,
              e_out_w + (size_t)i * Dc * Dc, e_out_b + i * Dc,
              e_ln2_w + i * Dc, e_ln2_b + i * Dc,
              e_fc1_w + (size_t)i * 4 * Dc * Dc, e_fc1_b + i * 4 * Dc,
              e_fc2_w + (size_t)i * Dc * 4 * Dc, e_fc2_b + i * Dc);

  // ---- encoder norm + decoder embed ----
  layernorm_k<<<Bc * Se, 32, 0, stream>>>(xe, lnt, e_nw, e_nb, Dc);
  gemm(lnt, Dc, 0, demb_w, Dc, 0, xd, DDc, 0, demb_b,
       nullptr, 0, 0, Bc * Se, DDc, Dc, 1, 1.0f, 0);

  // ---- scatter to full sequence + mask token + dec pos ----
  const int tfull = Bc * NPc * DDc;
  dec_init_k<<<cdiv(tfull, 256), 256, 0, stream>>>(xfull, mask_tok, dpos, tfull);
  const int tsc = Bc * 49 * DDc;
  dec_scatter_k<<<cdiv(tsc, 256), 256, 0, stream>>>(xd, unmasked, dpos, xfull, tsc);

  // ---- decoder ----
  for (int i = 0; i < DEPTHd; ++i)
    run_block(xfull, Sd, Scd, DDc, Hd, dhd,
              dc_ln1_w + i * DDc, dc_ln1_b + i * DDc,
              dc_qkv_w + (size_t)i * 3 * DDc * DDc, dc_qkv_b + i * 3 * DDc,
              dc_out_w + (size_t)i * DDc * DDc, dc_out_b + i * DDc,
              dc_ln2_w + i * DDc, dc_ln2_b + i * DDc,
              dc_fc1_w + (size_t)i * 4 * DDc * DDc, dc_fc1_b + i * 4 * DDc,
              dc_fc2_w + (size_t)i * DDc * 4 * DDc, dc_fc2_b + i * DDc);

  // ---- final norm + prediction head ----
  layernorm_k<<<Bc * Sd, 32, 0, stream>>>(xfull, lnt, dc_nw, dc_nb, DDc);
  gemm(lnt, DDc, 0, pred_w, DDc, 0, outp, 768, 0, pred_b,
       nullptr, 0, 0, Bc * Sd, 768, DDc, 1, 1.0f, 0);
}
